// EnhancedLiLTRelationExtraction_45002667327528
// MI455X (gfx1250) — compile-verified, matrix operand order checked by reading, not run on hardware
//
#include <hip/hip_runtime.h>

typedef __attribute__((ext_vector_type(16))) __bf16 v16bf;
typedef __attribute__((ext_vector_type(8)))  float  v8f;
typedef __attribute__((ext_vector_type(8)))  unsigned int v8u;
typedef __attribute__((ext_vector_type(2)))  __bf16 bf16x2;

constexpr int HD = 768;        // hidden
constexpr int SL = 512;        // seq len
constexpr int NB = 2;          // batch
constexpr int MR = NB * SL;    // 1024 rows
constexpr int NC = 2 * HD;     // combined proj cols (left|right) = 1536
constexpr int CH = HD / 2;     // chk hidden = 384

// K-offset within a 32-chunk for element e of a 16-bit A/B fragment (wave32).
// ISA 7.12.2: VGPR v<4 -> K base 2v, v>=4 -> 16+2(v-4); lanes>=16 add +8.
__device__ __host__ __forceinline__ int frag_koff(int e, int hi8) {
  int v = e >> 1;
  return ((v & 3) * 2) + ((v >> 2) * 16) + hi8 + (e & 1);
}

// packed bf16 relu(a+b) on a dword holding two bf16 (CDNA5 VOP3P)
__device__ __forceinline__ unsigned int relu_add_pk(unsigned int a, unsigned int b) {
  unsigned int s, r;
  asm("v_pk_add_bf16 %0, %1, %2" : "=v"(s) : "v"(a), "v"(b));
  asm("v_pk_max_num_bf16 %0, %1, %2" : "=v"(r) : "v"(s), "v"(0u));
  return r;
}

// ------------------------------------------------------------------ prep ----
__global__ void k_cvt_f32_bf16(const float* __restrict__ src,
                               __bf16* __restrict__ dst, int n) {
  int i = blockIdx.x * blockDim.x + threadIdx.x;
  int st = gridDim.x * blockDim.x;
  for (; i < n; i += st) dst[i] = (__bf16)src[i];
}

// Wcomb[h][n] (f32): n<768 -> wl+wd ; n>=768 -> wr-wd   (rel_w1 is (3H,H))
__global__ void k_combine(const float* __restrict__ rw1,
                          float* __restrict__ wcomb) {
  int i = blockIdx.x * blockDim.x + threadIdx.x;
  int st = gridDim.x * blockDim.x;
  const int total = HD * NC;
  for (; i < total; i += st) {
    int h = i / NC, n = i % NC;
    float v;
    if (n < HD) v = rw1[h * HD + n] + rw1[(2 * HD + h) * HD + n];
    else        v = rw1[(HD + h) * HD + (n - HD)] - rw1[(2 * HD + h) * HD + (n - HD)];
    wcomb[i] = v;
  }
}

// Pre-swizzle a K x Nvalid f32 weight matrix into WMMA B-fragment order:
// out dword index = ((nTile*chunks + c)*32 + lane)*8 + p, each dword = two bf16
// (K = c*32 + frag_koff(2p, hi8(lane)), K+1) for column n = nTile*16 + lane%16.
// Columns >= Nvalid are zero-padded.
__global__ void k_build_frag(const float* __restrict__ W, int Nvalid, int K,
                             int NT, unsigned int* __restrict__ out) {
  int chunks = K >> 5;
  int total = NT * chunks * 32 * 8;
  int i = blockIdx.x * blockDim.x + threadIdx.x;
  int st = gridDim.x * blockDim.x;
  for (; i < total; i += st) {
    int p    = i & 7;
    int lane = (i >> 3) & 31;
    int rest = i >> 8;
    int c    = rest % chunks;
    int nT   = rest / chunks;
    int n    = nT * 16 + (lane & 15);
    int hi8  = (lane & 16) ? 8 : 0;
    int k    = c * 32 + frag_koff(2 * p, hi8);
    __bf16 lo = (n < Nvalid) ? (__bf16)W[(size_t)k * Nvalid + n] : (__bf16)0.0f;
    __bf16 hi = (n < Nvalid) ? (__bf16)W[(size_t)(k + 1) * Nvalid + n] : (__bf16)0.0f;
    bf16x2 pk = {lo, hi};
    out[i] = __builtin_bit_cast(unsigned int, pk);
  }
}

// ------------------------------------------- generic bf16 WMMA wave GEMM ----
// Out[M][N] = opt_relu(A[M][K] * W[K][N] + bias[col<bias_len]), bf16 out.
// W supplied pre-swizzled in fragment order. One wave per 16x16 tile.
__global__ void __launch_bounds__(128) k_wmma_gemm_bf16(
    const __bf16* __restrict__ A, const unsigned int* __restrict__ Wfrag,
    const float* __restrict__ bias, int bias_len, int do_relu,
    int N, int K, __bf16* __restrict__ Out)
{
  int lane  = threadIdx.x & 31;
  int wave  = threadIdx.x >> 5;
  int nTile = blockIdx.x * 4 + wave;
  int mTile = blockIdx.y;
  if (nTile * 16 >= N) return;

  int mlo = lane & 15;
  int hi4 = (lane & 16) ? 4 : 0;          // dword offset of upper-K group
  int chunks = K >> 5;
  const unsigned int* arow =
      (const unsigned int*)(A + (size_t)(mTile * 16 + mlo) * K);
  const unsigned int* wbase = Wfrag + ((size_t)nTile * chunks * 32 + lane * 8);
  int col = nTile * 16 + mlo;

  v8f acc = {0.f, 0.f, 0.f, 0.f, 0.f, 0.f, 0.f, 0.f};
  for (int c = 0; c < chunks; ++c) {
    int dk = (c << 4) + hi4;
    uint4 a0 = *(const uint4*)(arow + dk);
    uint4 a1 = *(const uint4*)(arow + dk + 8);
    const unsigned int* wp = wbase + (size_t)c * 256;
    uint4 w0 = *(const uint4*)(wp);
    uint4 w1 = *(const uint4*)(wp + 4);
    v8u au = {a0.x, a0.y, a0.z, a0.w, a1.x, a1.y, a1.z, a1.w};
    v8u wu = {w0.x, w0.y, w0.z, w0.w, w1.x, w1.y, w1.z, w1.w};
    v16bf af = __builtin_bit_cast(v16bf, au);
    v16bf wf = __builtin_bit_cast(v16bf, wu);
    acc = __builtin_amdgcn_wmma_f32_16x16x32_bf16(false, af, false, wf,
                                                  (short)0, acc, false, false);
  }

  float bv = (bias != nullptr && col < bias_len) ? bias[col] : 0.0f;
  int hi8 = hi4 << 1;
#pragma unroll
  for (int r = 0; r < 8; ++r) {
    int row = mTile * 16 + r + hi8;
    float v = acc[r] + bv;
    if (do_relu) v = v > 0.f ? v : 0.f;
    Out[(size_t)row * N + col] = (__bf16)v;
  }
}

// ------------------------------------------------ pairwise relation head ----
// Per block: 16 i-rows x 16 j-rows for one batch. Wave w owns i-row i0+w and
// computes a 16(j) x 16(col; 3 valid) tile via 24 chained bf16 WMMAs.
// relu(L+R) built with packed bf16 VALU; w2 fragment pre-swizzled (L2-hot).
__global__ void __launch_bounds__(512) k_rel_pair(
    const __bf16* __restrict__ LR,            // [1024][1536] left|right
    const unsigned int* __restrict__ w2frag,  // [24][32][8] dwords
    const float* __restrict__ b2,             // [3]
    float* __restrict__ out)                  // [B][512][512][3]
{
  __shared__ __align__(16) __bf16 Ls[16][776];   // 1552B rows: 16B-aligned
  __shared__ __align__(16) __bf16 Rs[16][776];

  int b  = blockIdx.z;
  int i0 = blockIdx.y * 16;
  int j0 = blockIdx.x * 16;
  int tid = threadIdx.x;

  for (int idx = tid; idx < 16 * HD; idx += 512) {
    int r = idx / HD, k = idx - r * HD;
    Ls[r][k] = LR[(size_t)(b * SL + i0 + r) * NC + k];
    Rs[r][k] = LR[(size_t)(b * SL + j0 + r) * NC + HD + k];
  }
  __syncthreads();

  int lane = tid & 31;
  int w    = tid >> 5;                 // i-row within tile
  int m    = lane & 15;                // j-row within tile (A-fragment row)
  int hi4  = (lane & 16) ? 4 : 0;
  int ncol = lane & 15;                // output column of D fragment

  const unsigned int* lrow = (const unsigned int*)&Ls[w][0];
  const unsigned int* rrow = (const unsigned int*)&Rs[m][0];
  const unsigned int* wbase = w2frag + lane * 8;

  v8f acc = {0.f, 0.f, 0.f, 0.f, 0.f, 0.f, 0.f, 0.f};
  for (int c = 0; c < HD / 32; ++c) {
    int dk = (c << 4) + hi4;
    uint4 l0 = *(const uint4*)(lrow + dk);
    uint4 l1 = *(const uint4*)(lrow + dk + 8);
    uint4 r0 = *(const uint4*)(rrow + dk);
    uint4 r1 = *(const uint4*)(rrow + dk + 8);
    const unsigned int* wp = wbase + (size_t)c * 256;
    uint4 w0 = *(const uint4*)(wp);
    uint4 w1 = *(const uint4*)(wp + 4);

    v8u au;
    au[0] = relu_add_pk(l0.x, r0.x);
    au[1] = relu_add_pk(l0.y, r0.y);
    au[2] = relu_add_pk(l0.z, r0.z);
    au[3] = relu_add_pk(l0.w, r0.w);
    au[4] = relu_add_pk(l1.x, r1.x);
    au[5] = relu_add_pk(l1.y, r1.y);
    au[6] = relu_add_pk(l1.z, r1.z);
    au[7] = relu_add_pk(l1.w, r1.w);
    v8u wu = {w0.x, w0.y, w0.z, w0.w, w1.x, w1.y, w1.z, w1.w};
    v16bf af = __builtin_bit_cast(v16bf, au);
    v16bf wf = __builtin_bit_cast(v16bf, wu);
    acc = __builtin_amdgcn_wmma_f32_16x16x32_bf16(false, af, false, wf,
                                                  (short)0, acc, false, false);
  }

  if (ncol < 3) {
    float bias = b2[ncol];
    int i = i0 + w;
    int hi8 = hi4 << 1;
#pragma unroll
    for (int r = 0; r < 8; ++r) {
      int j = j0 + r + hi8;
      out[(((size_t)b * SL + i) * SL + j) * 3 + ncol] = acc[r] + bias;
    }
  }
}

// ----------------------------------------------------- checkmark logits -----
__global__ void __launch_bounds__(128) k_chk_logits(
    const __bf16* __restrict__ Hh,   // [1024][384] relu hidden (bf16)
    const float* __restrict__ w2,    // [384][4]
    const float* __restrict__ b2,    // [4]
    float* __restrict__ out)         // [1024][4]
{
  int row  = blockIdx.x;
  int o    = threadIdx.x >> 5;       // 4 waves -> 4 outputs
  int lane = threadIdx.x & 31;
  const __bf16* hr = Hh + (size_t)row * CH;
  float acc = 0.f;
  for (int j = lane; j < CH; j += 32) acc += (float)hr[j] * w2[j * 4 + o];
#pragma unroll
  for (int off = 16; off > 0; off >>= 1) acc += __shfl_down(acc, off, 32);
  if (lane == 0) out[row * 4 + o] = acc + b2[o];
}

// ------------------------------------------------------ CLS-row heads -------
__global__ void __launch_bounds__(256) k_cls_heads(
    const float* __restrict__ seq,
    const float* __restrict__ rsn_w1, const float* __restrict__ rsn_b1,
    const float* __restrict__ rsn_w2, const float* __restrict__ rsn_b2,
    const float* __restrict__ sol_w1, const float* __restrict__ sol_b1,
    const float* __restrict__ sol_w2, const float* __restrict__ sol_b2,
    float* __restrict__ out_rsn, float* __restrict__ out_sol)
{
  __shared__ float cls_s[HD];
  __shared__ float rsn_f[HD];
  __shared__ float sol_h[HD];
  int b = blockIdx.x;
  int t = threadIdx.x;
  for (int i = t; i < HD; i += 256) cls_s[i] = seq[(size_t)b * SL * HD + i];
  __syncthreads();

  for (int n = t; n < HD; n += 256) {
    float acc = rsn_b1[n];
    for (int k = 0; k < HD; ++k) acc += cls_s[k] * rsn_w1[k * HD + n];
    rsn_f[n] = acc > 0.f ? acc : 0.f;
  }
  __syncthreads();

  if (t < 8) {
    float acc = rsn_b2[t];
    for (int k = 0; k < HD; ++k) acc += rsn_f[k] * rsn_w2[k * 8 + t];
    out_rsn[b * 8 + t] = acc;
  }

  for (int n = t; n < HD; n += 256) {
    float acc = sol_b1[n];
    for (int k = 0; k < HD; ++k) acc += cls_s[k] * sol_w1[k * HD + n];
    for (int k = 0; k < HD; ++k) acc += rsn_f[k] * sol_w1[(HD + k) * HD + n];
    sol_h[n] = acc > 0.f ? acc : 0.f;
  }
  __syncthreads();

  if (t < 8) {
    float acc = sol_b2[t];
    for (int k = 0; k < HD; ++k) acc += sol_h[k] * sol_w2[k * 8 + t];
    out_sol[b * 8 + t] = acc;
  }
}

// ---------------------------------------------------------------- launch ----
extern "C" void kernel_launch(void* const* d_in, const int* in_sizes, int n_in,
                              void* d_out, int out_size, void* d_ws, size_t ws_size,
                              hipStream_t stream) {
  (void)in_sizes; (void)n_in; (void)out_size; (void)ws_size;
  const float* seq    = (const float*)d_in[0];
  const float* rel_w1 = (const float*)d_in[1];
  const float* rel_b1 = (const float*)d_in[2];
  const float* rel_w2 = (const float*)d_in[3];
  const float* rel_b2 = (const float*)d_in[4];
  const float* chk_w1 = (const float*)d_in[5];
  const float* chk_b1 = (const float*)d_in[6];
  const float* chk_w2 = (const float*)d_in[7];
  const float* chk_b2 = (const float*)d_in[8];
  const float* rsn_w1 = (const float*)d_in[9];
  const float* rsn_b1 = (const float*)d_in[10];
  const float* rsn_w2 = (const float*)d_in[11];
  const float* rsn_b2 = (const float*)d_in[12];
  const float* sol_w1 = (const float*)d_in[13];
  const float* sol_b1 = (const float*)d_in[14];
  const float* sol_w2 = (const float*)d_in[15];
  const float* sol_b2 = (const float*)d_in[16];

  char* ws = (char*)d_ws;
  size_t off = 0;
  auto alloc = [&](size_t bytes) -> void* {
    void* p = ws + off;
    off = (off + bytes + 255) & ~(size_t)255;
    return p;
  };
  __bf16* seqbf        = (__bf16*)alloc((size_t)MR * HD * 2);
  float* wcomb         = (float*)alloc((size_t)HD * NC * 4);
  unsigned int* pfrag  = (unsigned int*)alloc((size_t)(NC / 16) * (HD / 32) * 256 * 4);
  unsigned int* cfrag  = (unsigned int*)alloc((size_t)(CH / 16) * (HD / 32) * 256 * 4);
  unsigned int* w2frag = (unsigned int*)alloc((size_t)1 * (HD / 32) * 256 * 4);
  __bf16* LR           = (__bf16*)alloc((size_t)MR * NC * 2);
  __bf16* Hbf          = (__bf16*)alloc((size_t)MR * CH * 2);

  float* out     = (float*)d_out;
  float* out_rel = out;                                   // 2*512*512*3
  float* out_chk = out + (size_t)NB * SL * SL * 3;        // 2*512*4
  float* out_rsn = out_chk + (size_t)NB * SL * 4;         // 2*8
  float* out_sol = out_rsn + NB * 8;                      // 2*8

  k_cvt_f32_bf16<<<512, 256, 0, stream>>>(seq, seqbf, MR * HD);
  k_combine<<<512, 256, 0, stream>>>(rel_w1, wcomb);
  k_build_frag<<<512, 256, 0, stream>>>(wcomb, NC, HD, NC / 16, pfrag);
  k_build_frag<<<256, 256, 0, stream>>>(chk_w1, CH, HD, CH / 16, cfrag);
  k_build_frag<<<24, 256, 0, stream>>>(rel_w2, 3, HD, 1, w2frag);

  // left|right projection (rel_b1 folded into left half), bf16 out
  k_wmma_gemm_bf16<<<dim3(NC / 64, MR / 16), 128, 0, stream>>>(
      seqbf, pfrag, rel_b1, HD, 0, NC, HD, LR);
  // checkmark hidden: relu(seq@chk_w1 + b1), bf16 out
  k_wmma_gemm_bf16<<<dim3(CH / 64, MR / 16), 128, 0, stream>>>(
      seqbf, cfrag, chk_b1, CH, 1, CH, HD, Hbf);

  k_rel_pair<<<dim3(SL / 16, SL / 16, NB), 512, 0, stream>>>(
      LR, w2frag, rel_b2, out_rel);
  k_chk_logits<<<MR, 128, 0, stream>>>(Hbf, chk_w2, chk_b2, out_chk);
  k_cls_heads<<<NB, 256, 0, stream>>>(seq, rsn_w1, rsn_b1, rsn_w2, rsn_b2,
                                      sol_w1, sol_b1, sol_w2, sol_b2,
                                      out_rsn, out_sol);
}
